// MultiHeadAttention_1176821039969
// MI455X (gfx1250) — matrix operand order repaired and checked by reading, not compiled, not run on hardware
//
#include <hip/hip_runtime.h>
#include <hip/hip_bf16.h>
#include <stdint.h>

#define S_LEN   2048
#define D_MODEL 1024
#define N_HEADS 16
#define HEAD_DIM 64

typedef __attribute__((ext_vector_type(16))) __bf16 v16bf;
typedef __attribute__((ext_vector_type(8)))  float  v8f;

struct alignas(16) Frag32B { uint4 lo; uint4 hi; };

static __device__ __forceinline__ unsigned short f2bf(float f) {
  unsigned u = __builtin_bit_cast(unsigned, f);
  unsigned r = u + 0x7FFFu + ((u >> 16) & 1u);   // round-to-nearest-even
  return (unsigned short)(r >> 16);
}

static __device__ __forceinline__ v8f wmma_bf16(v16bf a, v16bf b, v8f c) {
  return __builtin_amdgcn_wmma_f32_16x16x32_bf16(false, a, false, b, (short)0, c, false, false);
}

// 16x32 bf16 A-fragment from row-major source: rows r0..r0+15, cols k0..k0+31.
// Lane L<16: row L, K {k0..k0+15}; lanes>=16: K {k0+16..k0+31}.
static __device__ __forceinline__ v16bf load_a_frag(const unsigned short* base,
                                                    int ldm, int r0, int k0, int lane) {
  const unsigned short* p = base + (size_t)(r0 + (lane & 15)) * ldm + k0 + ((lane >> 4) << 4);
  Frag32B f;
  f.lo = *(const uint4*)(p);
  f.hi = *(const uint4*)(p + 8);
  return __builtin_bit_cast(v16bf, f);
}

// 32x16 bf16 B-fragment. Source is row-major [N x K] (column n of B = row n of source).
// Lane L<16: col n0+L, K {k0..k0+15}; lanes>=16: K {k0+16..k0+31}. One 32B contiguous run.
static __device__ __forceinline__ v16bf load_b_frag(const unsigned short* base,
                                                    int ldm, int n0, int k0, int lane) {
  const unsigned short* p = base + (size_t)(n0 + (lane & 15)) * ldm + k0 + ((lane >> 4) << 4);
  Frag32B f;
  f.lo = *(const uint4*)(p);
  f.hi = *(const uint4*)(p + 8);
  return __builtin_bit_cast(v16bf, f);
}

// ---------------------------------------------------------------------------
// f32 -> bf16 cast
// ---------------------------------------------------------------------------
__global__ __launch_bounds__(256)
void cast_f32_bf16(const float* __restrict__ src, unsigned short* __restrict__ dst, int n) {
  int i = blockIdx.x * 256 + threadIdx.x;
  if (i < n) dst[i] = f2bf(src[i]);
}

// ---------------------------------------------------------------------------
// C[M,N] = A[M,K] @ B[N,K]^T   (bf16 in, f32 accumulate, bf16 or f32 out)
// Wave: 32x64 tile (2x4 WMMA), software-pipelined k-loop.
// Block: 8 waves as 4x2 -> 128 rows x 128 cols.
// ---------------------------------------------------------------------------
template <bool OUT_F32>
__global__ __launch_bounds__(256)
void gemm_bt_wmma(const unsigned short* __restrict__ A,
                  const unsigned short* __restrict__ Bw,
                  void* __restrict__ Cout, int N, int K) {
  const int lane = threadIdx.x & 31;
  const int wid  = threadIdx.x >> 5;
  const int r0 = blockIdx.y * 128 + (wid >> 1) * 32;
  const int c0 = blockIdx.x * 128 + (wid & 1) * 64;

  v8f acc[2][4] = {};

  v16bf a0 = load_a_frag(A, K, r0,      0, lane);
  v16bf a1 = load_a_frag(A, K, r0 + 16, 0, lane);
  v16bf b[4];
#pragma unroll
  for (int j = 0; j < 4; ++j) b[j] = load_b_frag(Bw, K, c0 + j * 16, 0, lane);

  for (int k0 = 0; k0 < K; k0 += 32) {
    v16bf na0, na1, nb[4];
    const int kn = k0 + 32;
    if (kn < K) {                       // issue next-slab loads before this slab's WMMAs
      na0 = load_a_frag(A, K, r0,      kn, lane);
      na1 = load_a_frag(A, K, r0 + 16, kn, lane);
#pragma unroll
      for (int j = 0; j < 4; ++j) nb[j] = load_b_frag(Bw, K, c0 + j * 16, kn, lane);
    } else {
      na0 = a0; na1 = a1;
#pragma unroll
      for (int j = 0; j < 4; ++j) nb[j] = b[j];
    }
#pragma unroll
    for (int j = 0; j < 4; ++j) {
      acc[0][j] = wmma_bf16(a0, b[j], acc[0][j]);
      acc[1][j] = wmma_bf16(a1, b[j], acc[1][j]);
    }
    a0 = na0; a1 = na1;
#pragma unroll
    for (int j = 0; j < 4; ++j) b[j] = nb[j];
  }

  const int rlo = (lane >> 4) << 3;
  const int cc  = lane & 15;
#pragma unroll
  for (int i = 0; i < 2; ++i)
#pragma unroll
    for (int j = 0; j < 4; ++j)
#pragma unroll
      for (int v = 0; v < 8; ++v) {
        size_t row = (size_t)(r0 + i * 16 + v + rlo);
        int    col = c0 + j * 16 + cc;
        float  val = acc[i][j][v];
        if (OUT_F32) ((float*)Cout)[row * N + col] = val;
        else         ((unsigned short*)Cout)[row * N + col] = f2bf(val);
      }
}

// ---------------------------------------------------------------------------
// Per-head V transpose: Vt[h][d][s] = V_head[h][s][d]
// ---------------------------------------------------------------------------
__global__ __launch_bounds__(256)
void transpose_v(const unsigned short* __restrict__ Vb, unsigned short* __restrict__ Vt) {
  unsigned idx = blockIdx.x * 256 + threadIdx.x;   // over H * HEAD_DIM * S = 2^21
  int s = idx & (S_LEN - 1);
  int d = (idx >> 11) & (HEAD_DIM - 1);
  int h = idx >> 17;
  size_t in = (size_t)(h * 128 + (s >> 4)) * D_MODEL + (size_t)((s & 15) * HEAD_DIM) + d;
  Vt[idx] = Vb[in];
}

// ---------------------------------------------------------------------------
// Flash attention, transposed formulation (S^T = K Q^T, O^T = V^T P).
// ---------------------------------------------------------------------------
struct KVFrags { v16bf ak[2][2]; v16bf av[4]; };

static __device__ __forceinline__ void load_kv(const unsigned short* Kh,
                                               const unsigned short* Vh,
                                               int j0, int lane, KVFrags& f) {
#pragma unroll
  for (int kt = 0; kt < 2; ++kt) {
    f.ak[kt][0] = load_a_frag(Kh, HEAD_DIM, j0 + kt * 16, 0,  lane);
    f.ak[kt][1] = load_a_frag(Kh, HEAD_DIM, j0 + kt * 16, 32, lane);
  }
#pragma unroll
  for (int nt = 0; nt < 4; ++nt) f.av[nt] = load_a_frag(Vh, S_LEN, nt * 16, j0, lane);
}

template <bool MASKED>
static __device__ __forceinline__ void attn_block(const KVFrags& f, v16bf bq0, v16bf bq1,
                                                  int j0, int q0, int cc, int rlo, int lane,
                                                  unsigned short* psw,
                                                  v8f o[4], float& m, float& l) {
  // S^T = K Q^T : 32 keys x 16 queries (k-dim = hd = 64)
  v8f sacc[2] = {};
#pragma unroll
  for (int kt = 0; kt < 2; ++kt) {
    sacc[kt] = wmma_bf16(f.ak[kt][0], bq0, sacc[kt]);
    sacc[kt] = wmma_bf16(f.ak[kt][1], bq1, sacc[kt]);
  }
  // scale (1/hd); causal mask only in the single diagonal block
  float s[2][8];
  float t = -1e30f;
#pragma unroll
  for (int kt = 0; kt < 2; ++kt)
#pragma unroll
    for (int v = 0; v < 8; ++v) {
      float sv = sacc[kt][v] * (1.0f / 64.0f);
      if (MASKED) {
        int key = j0 + kt * 16 + v + rlo;
        sv = (key > q0 + cc) ? -1e30f : sv;
      }
      s[kt][v] = sv;
      t = fmaxf(t, sv);
    }
  t = fmaxf(t, __shfl_xor(t, 16, 32));   // combine key-halves of this query
  float mn    = fmaxf(m, t);
  float alpha = __expf(m - mn);
  m = mn;

  float rs = 0.0f;
  uint4 pk[2];
#pragma unroll
  for (int kt = 0; kt < 2; ++kt) {
    float p[8];
#pragma unroll
    for (int v = 0; v < 8; ++v) { p[v] = __expf(s[kt][v] - mn); rs += p[v]; }
    pk[kt].x = (unsigned)f2bf(p[0]) | ((unsigned)f2bf(p[1]) << 16);
    pk[kt].y = (unsigned)f2bf(p[2]) | ((unsigned)f2bf(p[3]) << 16);
    pk[kt].z = (unsigned)f2bf(p[4]) | ((unsigned)f2bf(p[5]) << 16);
    pk[kt].w = (unsigned)f2bf(p[6]) | ((unsigned)f2bf(p[7]) << 16);
  }
  rs += __shfl_xor(rs, 16, 32);
  l = l * alpha + rs;

  // stage P row-major [query][key]: two 16B ds_store_b128 per lane
#pragma unroll
  for (int kt = 0; kt < 2; ++kt)
    *(uint4*)(psw + cc * 32 + kt * 16 + rlo) = pk[kt];
  v16bf bp = load_b_frag(psw, 32, 0, 0, lane);

  // O^T = alpha * O^T + V^T P   (alpha: uniform per-lane scalar)
#pragma unroll
  for (int nt = 0; nt < 4; ++nt) {
#pragma unroll
    for (int v = 0; v < 8; ++v) o[nt][v] *= alpha;
    o[nt] = wmma_bf16(f.av[nt], bp, o[nt]);
  }
}

__global__ __launch_bounds__(256)
void flash_attn_wmma(const unsigned short* __restrict__ Qb,
                     const unsigned short* __restrict__ Kb,
                     const unsigned short* __restrict__ Vt,
                     unsigned short* __restrict__ Actx) {
  __shared__ __align__(16) unsigned short ps[8][16][32];
  const int lane = threadIdx.x & 31;
  const int wid  = threadIdx.x >> 5;
  const int h  = blockIdx.y;
  const int q0 = blockIdx.x * 128 + wid * 16;
  const int cc  = lane & 15;
  const int rlo = (lane >> 4) << 3;
  unsigned short* psw = &ps[wid][0][0];

  const unsigned short* Qh = Qb + (size_t)h * S_LEN * HEAD_DIM;
  const unsigned short* Kh = Kb + (size_t)h * S_LEN * HEAD_DIM;
  const unsigned short* Vh = Vt + (size_t)h * HEAD_DIM * S_LEN;

  v16bf bq0 = load_b_frag(Qh, HEAD_DIM, q0, 0,  lane);
  v16bf bq1 = load_b_frag(Qh, HEAD_DIM, q0, 32, lane);

  v8f o[4] = {};
  float m = -1e30f, l = 0.0f;

  // Exactly ONE partially-masked (diagonal) block per wave: jm = (q0+15) & ~31.
  // Blocks j0 < jm are fully unmasked (all keys <= all queries).
  const int jm = (q0 + 15) & ~31;

  KVFrags A, B;                         // ping-pong fragment buffers
  load_kv(Kh, Vh, 0, lane, A);
  if (jm >= 32) {
    int j0 = 0;
    for (; j0 + 64 <= jm; j0 += 64) {   // 2x-unrolled, software-pipelined main loop
      load_kv(Kh, Vh, j0 + 32, lane, B);
      attn_block<false>(A, bq0, bq1, j0, q0, cc, rlo, lane, psw, o, m, l);
      load_kv(Kh, Vh, j0 + 64, lane, A);
      attn_block<false>(B, bq0, bq1, j0 + 32, q0, cc, rlo, lane, psw, o, m, l);
    }
    if (j0 < jm) {                      // odd remainder: one main block + masked tail
      load_kv(Kh, Vh, j0 + 32, lane, B);
      attn_block<false>(A, bq0, bq1, j0, q0, cc, rlo, lane, psw, o, m, l);
      attn_block<true >(B, bq0, bq1, jm, q0, cc, rlo, lane, psw, o, m, l);
    } else {
      attn_block<true >(A, bq0, bq1, jm, q0, cc, rlo, lane, psw, o, m, l);
    }
  } else {
    attn_block<true>(A, bq0, bq1, 0, q0, cc, rlo, lane, psw, o, m, l);
  }

  // normalize; per-lane outputs along v are contiguous in Actx -> 16B packed stores
  float invl = 1.0f / l;
#pragma unroll
  for (int nt = 0; nt < 4; ++nt) {
    uint4 w;
    w.x = (unsigned)f2bf(o[nt][0] * invl) | ((unsigned)f2bf(o[nt][1] * invl) << 16);
    w.y = (unsigned)f2bf(o[nt][2] * invl) | ((unsigned)f2bf(o[nt][3] * invl) << 16);
    w.z = (unsigned)f2bf(o[nt][4] * invl) | ((unsigned)f2bf(o[nt][5] * invl) << 16);
    w.w = (unsigned)f2bf(o[nt][6] * invl) | ((unsigned)f2bf(o[nt][7] * invl) << 16);
    *(uint4*)&Actx[(size_t)(q0 + cc) * D_MODEL + h * HEAD_DIM + nt * 16 + rlo] = w;
  }
}

// ---------------------------------------------------------------------------
extern "C" void kernel_launch(void* const* d_in, const int* in_sizes, int n_in,
                              void* d_out, int out_size, void* d_ws, size_t ws_size,
                              hipStream_t stream) {
  const float* x  = (const float*)d_in[0];
  const float* Wq = (const float*)d_in[1];
  const float* Wk = (const float*)d_in[2];
  const float* Wv = (const float*)d_in[3];
  const float* Wo = (const float*)d_in[4];

  char* ws = (char*)d_ws;
  const size_t MB = 1024u * 1024u;
  unsigned short* xb  = (unsigned short*)(ws + 0);        // 4 MB  [2048,1024] bf16
  unsigned short* wqb = (unsigned short*)(ws + 4  * MB);  // 2 MB each
  unsigned short* wkb = (unsigned short*)(ws + 6  * MB);
  unsigned short* wvb = (unsigned short*)(ws + 8  * MB);
  unsigned short* wob = (unsigned short*)(ws + 10 * MB);
  unsigned short* Qb  = (unsigned short*)(ws + 12 * MB);  // 4 MB each
  unsigned short* Kb  = (unsigned short*)(ws + 16 * MB);
  unsigned short* Vb  = (unsigned short*)(ws + 20 * MB);
  unsigned short* Vt  = (unsigned short*)(ws + 24 * MB);  // [H][64][2048]
  unsigned short* Ab  = (unsigned short*)(ws + 28 * MB);  // attention context, bf16

  const int SD = S_LEN * D_MODEL;     // 2M
  const int DD = D_MODEL * D_MODEL;   // 1M
  cast_f32_bf16<<<SD / 256, 256, 0, stream>>>(x,  xb,  SD);
  cast_f32_bf16<<<DD / 256, 256, 0, stream>>>(Wq, wqb, DD);
  cast_f32_bf16<<<DD / 256, 256, 0, stream>>>(Wk, wkb, DD);
  cast_f32_bf16<<<DD / 256, 256, 0, stream>>>(Wv, wvb, DD);
  cast_f32_bf16<<<DD / 256, 256, 0, stream>>>(Wo, wob, DD);

  dim3 ggrid(D_MODEL / 128, S_LEN / 128);   // (8, 16)
  gemm_bt_wmma<false><<<ggrid, 256, 0, stream>>>(xb, wqb, Qb, D_MODEL, D_MODEL);
  gemm_bt_wmma<false><<<ggrid, 256, 0, stream>>>(xb, wkb, Kb, D_MODEL, D_MODEL);
  gemm_bt_wmma<false><<<ggrid, 256, 0, stream>>>(xb, wvb, Vb, D_MODEL, D_MODEL);

  transpose_v<<<(N_HEADS * HEAD_DIM * S_LEN) / 256, 256, 0, stream>>>(Vb, Vt);

  flash_attn_wmma<<<dim3(S_LEN / 128, N_HEADS), 256, 0, stream>>>(Qb, Kb, Vt, Ab);

  gemm_bt_wmma<true><<<ggrid, 256, 0, stream>>>(Ab, wob, (float*)d_out, D_MODEL, D_MODEL);
}
